// Model_7310034338115
// MI455X (gfx1250) — matrix operand order repaired and checked by reading, not compiled, not run on hardware
//
#include <hip/hip_runtime.h>
#include <hip/hip_bf16.h>
#include <math.h>

#define B_ 4
#define S_ 1024
#define D_ 192
#define H_ 8
#define DK_ 64
#define IN_ 512   // H_*DK_
#define WSEG 98304  // 192*512 == 512*192: element count of each weight matrix

typedef __attribute__((ext_vector_type(16))) __bf16 v16bf;
typedef __attribute__((ext_vector_type(8)))  float  v8f;

// native f32 -> bf16 convert (single v_cvt instruction on gfx1250)
__device__ __forceinline__ unsigned short f2bf(float f) {
  union { __bf16 h; unsigned short s; } u;
  u.h = (__bf16)f;
  return u.s;
}

__device__ __forceinline__ v8f vzero() {
  v8f z = {0.f, 0.f, 0.f, 0.f, 0.f, 0.f, 0.f, 0.f};
  return z;
}

__device__ __forceinline__ v8f wmma_bf16(v16bf a, v16bf b, v8f c) {
  // (neg_a, A, neg_b, B, c_mod, C, reuse_a, reuse_b) -> v_wmma_f32_16x16x32_bf16
  return __builtin_amdgcn_wmma_f32_16x16x32_bf16(false, a, false, b, (short)0, c,
                                                 false, false);
}

union Frag { v16bf v; unsigned u[8]; unsigned short s[16]; };

// A operand: 16x32 bf16 tile from row-major bf16 (ushort) memory, stride ld elems.
// ISA layout: lane<16 -> row=lane, K={0..7,16..23}; lane>=16 -> row=lane-16, K={8..15,24..31}
__device__ __forceinline__ v16bf load_a_frag(const unsigned short* base, int ld) {
  const int lane = threadIdx.x & 31;
  const int row  = lane & 15;
  const int klo  = (lane >> 4) << 3;   // 0 or 8
  const unsigned short* p = base + row * ld;
  Frag f;
#pragma unroll
  for (int i = 0; i < 4; ++i) {
    f.u[i]     = *(const unsigned*)(p + klo + 2 * i);
    f.u[4 + i] = *(const unsigned*)(p + 16 + klo + 2 * i);
  }
  return f.v;
}

// B operand: 32x16 bf16 tile from row-major [K x N] bf16 memory, stride ld elems.
// ISA layout: lane<16 -> col=lane, K={0..7,16..23}; lane>=16 -> col=lane-16, K={8..15,24..31}
__device__ __forceinline__ v16bf load_b_frag_bf(const unsigned short* base, int ld) {
  const int lane = threadIdx.x & 31;
  const int klo  = (lane >> 4) << 3;
  const unsigned short* p = base + (lane & 15);
  Frag f;
#pragma unroll
  for (int i = 0; i < 8; ++i) {
    f.s[i]     = p[(klo + i) * ld];
    f.s[8 + i] = p[(16 + klo + i) * ld];
  }
  return f.v;
}

// ---------------------------------------------------------------- weight convert
// One-time f32 -> bf16 conversion of Wq|Wk|Wv|Wo into one packed buffer.
__global__ __launch_bounds__(256) void wcvt_kernel(
    const float* __restrict__ Wq, const float* __restrict__ Wk,
    const float* __restrict__ Wv, const float* __restrict__ Wo,
    unsigned* __restrict__ dst) {
  const int i = blockIdx.x * 256 + threadIdx.x;   // each thread: 2 elements
  const int e = i * 2;
  if (e >= 4 * WSEG) return;
  const int seg = e / WSEG, off = e % WSEG;
  const float* src = (seg == 0) ? Wq : (seg == 1) ? Wk : (seg == 2) ? Wv : Wo;
  const unsigned lo = f2bf(src[off]);
  const unsigned hi = f2bf(src[off + 1]);
  dst[i] = lo | (hi << 16);
}

// ---------------------------------------------------------------- bias table
// bias[h][rel+1023] = rel_emb[bucket(rel)][h], HF T5 bidirectional bucketing.
__global__ __launch_bounds__(256) void bias_table_kernel(
    const float* __restrict__ rel_emb, float* __restrict__ tbl) {
  int i = blockIdx.x * 256 + threadIdx.x;
  const int n = H_ * (2 * S_ - 1);
  if (i >= n) return;
  int h   = i / (2 * S_ - 1);
  int rel = (i % (2 * S_ - 1)) - (S_ - 1);
  int bucket = (rel > 0) ? 16 : 0;
  int rp = rel < 0 ? -rel : rel;
  if (rp < 8) {
    bucket += rp;
  } else {
    int large = 8 + (int)(__logf((float)rp * 0.125f) / __logf(16.0f) * 8.0f);
    bucket += (large < 15) ? large : 15;
  }
  tbl[i] = rel_emb[bucket * H_ + h];
}

// ---------------------------------------------------------------- QKV GEMM
// x[4096,192] @ W[192,512] -> Q/V as [B,H,S,DK] bf16, K transposed [B,H,DK,S] bf16.
__global__ __launch_bounds__(128) void qkv_kernel(
    const float* __restrict__ x, const unsigned short* __restrict__ Wall,
    unsigned short* __restrict__ Qb, unsigned short* __restrict__ Ktr,
    unsigned short* __restrict__ Vb) {
  __shared__ unsigned short xs[16 * D_];
  const int m0 = blockIdx.x * 16;
  const int w  = blockIdx.y;                         // 0=Q 1=K 2=V
  const unsigned short* W = Wall + (size_t)w * WSEG; // bf16 weights

  for (int i = threadIdx.x; i < 16 * D_; i += 128)
    xs[i] = f2bf(x[(m0 + i / D_) * D_ + (i % D_)]);
  __syncthreads();

  const int wave = threadIdx.x >> 5;
  const int lane = threadIdx.x & 31;
  const int n0   = wave * 128;                       // 8 n-tiles per wave
  v8f acc[8];
#pragma unroll
  for (int t = 0; t < 8; ++t) acc[t] = vzero();

  for (int kk = 0; kk < D_ / 32; ++kk) {             // 6 K-steps
    v16bf a = load_a_frag(xs + kk * 32, D_);
#pragma unroll
    for (int t = 0; t < 8; ++t) {
      v16bf bm = load_b_frag_bf(W + (kk * 32) * IN_ + n0 + t * 16, IN_);
      acc[t] = wmma_bf16(a, bm, acc[t]);
    }
  }

  const int rbase = (lane < 16) ? 0 : 8;
  const int coll  = lane & 15;
#pragma unroll
  for (int t = 0; t < 8; ++t) {
    const int n = n0 + t * 16 + coll;
    const int h = n >> 6, dk = n & 63;
#pragma unroll
    for (int r = 0; r < 8; ++r) {
      const int m = m0 + rbase + r;
      const int bb = m >> 10, s = m & (S_ - 1);
      unsigned short val = f2bf(acc[t][r]);
      if (w == 0)      Qb [((bb * H_ + h) * S_  + s ) * DK_ + dk] = val;
      else if (w == 1) Ktr[((bb * H_ + h) * DK_ + dk) * S_  + s ] = val;
      else             Vb [((bb * H_ + h) * S_  + s ) * DK_ + dk] = val;
    }
  }
}

// ---------------------------------------------------------------- flash attention
// One wave = one 16-query tile; block = 8 waves = 128 queries of one (b,h).
__global__ __launch_bounds__(256) void attn_kernel(
    const unsigned short* __restrict__ Qb, const unsigned short* __restrict__ Ktr,
    const unsigned short* __restrict__ Vb, const float* __restrict__ mask,
    const float* __restrict__ biasTbl, unsigned short* __restrict__ Ao) {
  __shared__ float maskLds[S_];
  __shared__ float biasLds[2 * S_ - 1];
  __shared__ unsigned short probLds[8][16 * 32];

  const int h  = blockIdx.y;
  const int bb = blockIdx.z;
  for (int i = threadIdx.x; i < S_; i += 256) maskLds[i] = mask[bb * S_ + i];
  for (int i = threadIdx.x; i < 2 * S_ - 1; i += 256)
    biasLds[i] = biasTbl[h * (2 * S_ - 1) + i];
  __syncthreads();

  const int wave = threadIdx.x >> 5;
  const int lane = threadIdx.x & 31;
  const int q0   = blockIdx.x * 128 + wave * 16;
  const unsigned short* Qh = Qb  + (size_t)(bb * H_ + h) * S_  * DK_;
  const unsigned short* Kh = Ktr + (size_t)(bb * H_ + h) * DK_ * S_;
  const unsigned short* Vh = Vb  + (size_t)(bb * H_ + h) * S_  * DK_;

  const v16bf qa0 = load_a_frag(Qh + q0 * DK_,      DK_);   // K dims 0..31
  const v16bf qa1 = load_a_frag(Qh + q0 * DK_ + 32, DK_);   // K dims 32..63

  v8f o0 = vzero(), o1 = vzero(), o2 = vzero(), o3 = vzero();
  float m[8], l[8];
#pragma unroll
  for (int r = 0; r < 8; ++r) { m[r] = -1e30f; l[r] = 0.f; }
  const int rbase = (lane < 16) ? 0 : 8;
  const int coll  = lane & 15;
  unsigned short* pl = &probLds[wave][0];

  for (int kc = 0; kc < S_ / 32; ++kc) {
    const int k0 = kc * 32;
    // prefetch next V chunk (CDNA5 global_prefetch_b8)
    if (kc + 1 < S_ / 32)
      __builtin_prefetch((const void*)(Vh + (k0 + 32) * DK_ + lane * DK_), 0, 1);

    v8f st[2];
#pragma unroll
    for (int t = 0; t < 2; ++t) {
      v8f a = vzero();
      a = wmma_bf16(qa0, load_b_frag_bf(Kh + k0 + t * 16,            S_), a);
      a = wmma_bf16(qa1, load_b_frag_bf(Kh + 32 * S_ + k0 + t * 16,  S_), a);
      const int kpos = k0 + t * 16 + coll;
      const float mval = maskLds[kpos];
#pragma unroll
      for (int r = 0; r < 8; ++r) {
        const int qpos = q0 + rbase + r;
        a[r] += biasLds[kpos - qpos + (S_ - 1)] + mval;
      }
      st[t] = a;
    }

    // row max across both key tiles + 16-lane horizontal reduce, online update
    float alpha[8];
#pragma unroll
    for (int r = 0; r < 8; ++r) {
      float t = fmaxf(st[0][r], st[1][r]);
      t = fmaxf(t, __shfl_xor(t, 1));
      t = fmaxf(t, __shfl_xor(t, 2));
      t = fmaxf(t, __shfl_xor(t, 4));
      t = fmaxf(t, __shfl_xor(t, 8));
      const float mn = fmaxf(m[r], t);
      alpha[r] = __expf(m[r] - mn);
      m[r] = mn;
      l[r] *= alpha[r];
      o0[r] *= alpha[r]; o1[r] *= alpha[r]; o2[r] *= alpha[r]; o3[r] *= alpha[r];
    }

    // probabilities -> row sums -> per-wave LDS staging (16x32 bf16)
#pragma unroll
    for (int t = 0; t < 2; ++t) {
#pragma unroll
      for (int r = 0; r < 8; ++r) {
        const float p = __expf(st[t][r] - m[r]);
        float s = p;
        s += __shfl_xor(s, 1);
        s += __shfl_xor(s, 2);
        s += __shfl_xor(s, 4);
        s += __shfl_xor(s, 8);
        l[r] += s;
        pl[(rbase + r) * 32 + t * 16 + coll] = f2bf(p);
      }
    }

    // P(16x32) @ V(32x64): re-read probs in A-operand layout, 4 WMMAs
    const v16bf pa = load_a_frag(pl, 32);
    o0 = wmma_bf16(pa, load_b_frag_bf(Vh + k0 * DK_ +  0, DK_), o0);
    o1 = wmma_bf16(pa, load_b_frag_bf(Vh + k0 * DK_ + 16, DK_), o1);
    o2 = wmma_bf16(pa, load_b_frag_bf(Vh + k0 * DK_ + 32, DK_), o2);
    o3 = wmma_bf16(pa, load_b_frag_bf(Vh + k0 * DK_ + 48, DK_), o3);
  }

  // normalize and write attn output [B, S, H*DK] bf16
#pragma unroll
  for (int r = 0; r < 8; ++r) {
    const int q = q0 + rbase + r;
    const float inv = 1.f / l[r];
    unsigned short* dst = Ao + ((size_t)bb * S_ + q) * IN_ + h * DK_;
    dst[coll]      = f2bf(o0[r] * inv);
    dst[16 + coll] = f2bf(o1[r] * inv);
    dst[32 + coll] = f2bf(o2[r] * inv);
    dst[48 + coll] = f2bf(o3[r] * inv);
  }
}

// ---------------------------------------------------------------- output projection
// Ao[4096,512] bf16 @ Wo[512,192] bf16 -> out[4096,192] f32
__global__ __launch_bounds__(128) void proj_kernel(
    const unsigned short* __restrict__ Ao, const unsigned short* __restrict__ Wob,
    float* __restrict__ out) {
  __shared__ unsigned short as[16 * IN_];
  const int m0 = blockIdx.x * 16;
  const unsigned* src = (const unsigned*)(Ao + (size_t)m0 * IN_);
  unsigned* dst = (unsigned*)as;
  for (int i = threadIdx.x; i < 16 * IN_ / 2; i += 128) dst[i] = src[i];
  __syncthreads();

  const int wave = threadIdx.x >> 5;
  const int lane = threadIdx.x & 31;
  const int n0   = wave * 48;                       // 3 n-tiles per wave
  v8f acc[3] = {vzero(), vzero(), vzero()};

  for (int kk = 0; kk < IN_ / 32; ++kk) {           // 16 K-steps
    v16bf a = load_a_frag(as + kk * 32, IN_);
#pragma unroll
    for (int t = 0; t < 3; ++t)
      acc[t] = wmma_bf16(a, load_b_frag_bf(Wob + kk * 32 * D_ + n0 + t * 16, D_),
                         acc[t]);
  }

  const int rbase = (lane < 16) ? 0 : 8;
  const int coll  = lane & 15;
#pragma unroll
  for (int t = 0; t < 3; ++t)
#pragma unroll
    for (int r = 0; r < 8; ++r)
      out[(m0 + rbase + r) * D_ + n0 + t * 16 + coll] = acc[t][r];
}

// ----------------------------------------------------------------
extern "C" void kernel_launch(void* const* d_in, const int* in_sizes, int n_in,
                              void* d_out, int out_size, void* d_ws, size_t ws_size,
                              hipStream_t stream) {
  (void)in_sizes; (void)n_in; (void)out_size; (void)ws_size;
  const float* x       = (const float*)d_in[0];
  const float* mask    = (const float*)d_in[1];
  const float* Wq      = (const float*)d_in[2];
  const float* Wk      = (const float*)d_in[3];
  const float* Wv      = (const float*)d_in[4];
  const float* Wo      = (const float*)d_in[5];
  const float* rel_emb = (const float*)d_in[6];

  char* ws = (char*)d_ws;
  size_t off = 0;
  const size_t qkv_bytes = (size_t)B_ * H_ * S_ * DK_ * sizeof(unsigned short);
  unsigned short* Qb  = (unsigned short*)(ws + off); off += qkv_bytes;
  unsigned short* Ktr = (unsigned short*)(ws + off); off += qkv_bytes;
  unsigned short* Vb  = (unsigned short*)(ws + off); off += qkv_bytes;
  unsigned short* Ao  = (unsigned short*)(ws + off); off += (size_t)B_ * S_ * IN_ * sizeof(unsigned short);
  float* biasTbl      = (float*)(ws + off);          off += (size_t)H_ * (2 * S_ - 1) * sizeof(float);
  unsigned short* Wb  = (unsigned short*)(ws + off); off += (size_t)4 * WSEG * sizeof(unsigned short);

  wcvt_kernel<<<(4 * WSEG / 2 + 255) / 256, 256, 0, stream>>>(Wq, Wk, Wv, Wo,
                                                              (unsigned*)Wb);
  bias_table_kernel<<<(H_ * (2 * S_ - 1) + 255) / 256, 256, 0, stream>>>(rel_emb, biasTbl);
  qkv_kernel<<<dim3(B_ * S_ / 16, 3), 128, 0, stream>>>(x, Wb, Qb, Ktr, Vb);
  attn_kernel<<<dim3(S_ / 128, H_, B_), 256, 0, stream>>>(Qb, Ktr, Vb, mask, biasTbl, Ao);
  proj_kernel<<<B_ * S_ / 16, 128, 0, stream>>>(Ao, Wb + (size_t)3 * WSEG, (float*)d_out);
}
